// RPEMultiHeadAttention_23725399343536
// MI455X (gfx1250) — matrix operand order repaired and checked by reading, not compiled, not run on hardware
//
#include <hip/hip_runtime.h>
#include <hip/hip_bf16.h>

typedef __attribute__((ext_vector_type(16))) __bf16 v16bf;
typedef __attribute__((ext_vector_type(8)))  float  v8f;
typedef __attribute__((ext_vector_type(4)))  float  v4f;

#define WMMA_BF16(a, b, c) \
  __builtin_amdgcn_wmma_f32_16x16x32_bf16(false, (a), false, (b), (short)0, (c), false, false)

// A-matrix (16x32 bf16) element i -> K index within the 32-chunk (ISA 7.12.2):
// elements 0..7  -> K = half*8 + i        (one 8-elem contiguous run)
// elements 8..15 -> K = 16 + half*8 + i-8 (second 8-elem contiguous run)
__device__ __forceinline__ int amap(int i, int half) {
  return (i < 8) ? (half * 8 + i) : (16 + half * 8 + (i - 8));
}

// Load a bf16 A-tile chunk: two contiguous 8-bf16 runs -> two b128 loads.
__device__ __forceinline__ v16bf load_a_bf16(const __bf16* row, int kc, int half) {
  union { uint4 u[2]; v16bf v; } t;
  t.u[0] = *(const uint4*)(row + kc + half * 16);
  t.u[1] = *(const uint4*)(row + kc + half * 16 + 8);
  return t.v;
}

// Load 16 contiguous bf16 (B-operand run) -> two b128 loads.
__device__ __forceinline__ v16bf load_b16x16(const __bf16* p) {
  union { uint4 u[2]; v16bf v; } t;
  const uint4* q = (const uint4*)p;
  t.u[0] = q[0]; t.u[1] = q[1];
  return t.v;
}

// f32 A-tile chunk -> 4x b128 loads + pack to bf16 (optionally non-temporal).
__device__ __forceinline__ v16bf cvt_a_f32(const float* row, int kc, int half) {
  const v4f* p0 = (const v4f*)(row + kc + half * 16);
  const v4f* p1 = (const v4f*)(row + kc + half * 16 + 8);
  v4f x0 = p0[0], x1 = p0[1], y0 = p1[0], y1 = p1[1];
  v16bf a;
  a[0]  = (__bf16)x0[0]; a[1]  = (__bf16)x0[1]; a[2]  = (__bf16)x0[2]; a[3]  = (__bf16)x0[3];
  a[4]  = (__bf16)x1[0]; a[5]  = (__bf16)x1[1]; a[6]  = (__bf16)x1[2]; a[7]  = (__bf16)x1[3];
  a[8]  = (__bf16)y0[0]; a[9]  = (__bf16)y0[1]; a[10] = (__bf16)y0[2]; a[11] = (__bf16)y0[3];
  a[12] = (__bf16)y1[0]; a[13] = (__bf16)y1[1]; a[14] = (__bf16)y1[2]; a[15] = (__bf16)y1[3];
  return a;
}
__device__ __forceinline__ v16bf cvt_a_f32_nt(const float* row, int kc, int half) {
  const v4f* p0 = (const v4f*)(row + kc + half * 16);
  const v4f* p1 = (const v4f*)(row + kc + half * 16 + 8);
  v4f x0 = __builtin_nontemporal_load(p0);
  v4f x1 = __builtin_nontemporal_load(p0 + 1);
  v4f y0 = __builtin_nontemporal_load(p1);
  v4f y1 = __builtin_nontemporal_load(p1 + 1);
  v16bf a;
  a[0]  = (__bf16)x0[0]; a[1]  = (__bf16)x0[1]; a[2]  = (__bf16)x0[2]; a[3]  = (__bf16)x0[3];
  a[4]  = (__bf16)x1[0]; a[5]  = (__bf16)x1[1]; a[6]  = (__bf16)x1[2]; a[7]  = (__bf16)x1[3];
  a[8]  = (__bf16)y0[0]; a[9]  = (__bf16)y0[1]; a[10] = (__bf16)y0[2]; a[11] = (__bf16)y0[3];
  a[12] = (__bf16)y1[0]; a[13] = (__bf16)y1[1]; a[14] = (__bf16)y1[2]; a[15] = (__bf16)y1[3];
  return a;
}

// ---------------------------------------------------------------------------
// 1) Q/K/V projections: out = X @ W^T + b, bf16 output (512x256 each)
// ---------------------------------------------------------------------------
__global__ __launch_bounds__(32) void proj_qkv_kernel(
    const float* __restrict__ Xq, const float* __restrict__ Xk, const float* __restrict__ Xv,
    const float* __restrict__ Wq, const float* __restrict__ Wk, const float* __restrict__ Wv,
    const float* __restrict__ bq, const float* __restrict__ bk, const float* __restrict__ bv,
    __bf16* __restrict__ Qbf, __bf16* __restrict__ Kbf, __bf16* __restrict__ Vbf)
{
  const float* X; const float* W; const float* bias; __bf16* out;
  if (blockIdx.y == 0)      { X = Xq; W = Wq; bias = bq; out = Qbf; }
  else if (blockIdx.y == 1) { X = Xk; W = Wk; bias = bk; out = Kbf; }
  else                      { X = Xv; W = Wv; bias = bv; out = Vbf; }
  int tile = blockIdx.x;
  int tn = tile >> 4, tj = tile & 15;
  int lane = threadIdx.x, lr = lane & 15, half = lane >> 4;
  int row = tn * 16 + lr;   // input row (A-matrix M)
  int col = tj * 16 + lr;   // output col (B-matrix N = W row, since out = X @ W^T)
  const float* Xrow = X + (size_t)row * 256;
  const float* Wrow = W + (size_t)col * 256;
  v8f acc = {};
  for (int kc = 0; kc < 256; kc += 32) {
    v16bf a = cvt_a_f32(Xrow, kc, half);
    // B element i: K = half*16 + i -> 16 contiguous f32 of W row
    const v4f* wp = (const v4f*)(Wrow + kc + half * 16);
    v4f w0 = wp[0], w1 = wp[1], w2 = wp[2], w3 = wp[3];
    v16bf b;
    b[0]  = (__bf16)w0[0]; b[1]  = (__bf16)w0[1]; b[2]  = (__bf16)w0[2]; b[3]  = (__bf16)w0[3];
    b[4]  = (__bf16)w1[0]; b[5]  = (__bf16)w1[1]; b[6]  = (__bf16)w1[2]; b[7]  = (__bf16)w1[3];
    b[8]  = (__bf16)w2[0]; b[9]  = (__bf16)w2[1]; b[10] = (__bf16)w2[2]; b[11] = (__bf16)w2[3];
    b[12] = (__bf16)w3[0]; b[13] = (__bf16)w3[1]; b[14] = (__bf16)w3[2]; b[15] = (__bf16)w3[3];
    acc = WMMA_BF16(a, b, acc);
  }
#pragma unroll
  for (int r = 0; r < 8; ++r) {
    int n = tn * 16 + r + half * 8;
    out[n * 256 + col] = (__bf16)(acc[r] + bias[col]);
  }
}

// ---------------------------------------------------------------------------
// 2a) qp: QPE[n][h][k] = sum_c Q[n, h*32+c] * Wp[h*32+c, k], k < 256
// ---------------------------------------------------------------------------
__global__ __launch_bounds__(32) void qp_kernel(
    const __bf16* __restrict__ Qbf, const float* __restrict__ Wp,
    __bf16* __restrict__ QPE)
{
  int id = blockIdx.x;          // 8 * 32 * 16 tiles
  int h = id >> 9;
  int rem = id & 511;
  int tn = rem >> 4, tk = rem & 15;
  int lane = threadIdx.x, lr = lane & 15, half = lane >> 4;
  int row  = tn * 16 + lr;
  int kcol = tk * 16 + lr;
  v16bf a = load_a_bf16(Qbf + (size_t)row * 256 + h * 32, 0, half);
  v16bf b;
#pragma unroll
  for (int i = 0; i < 16; ++i) b[i] = (__bf16)Wp[(h * 32 + half * 16 + i) * 256 + kcol];
  v8f acc = {};
  acc = WMMA_BF16(a, b, acc);
#pragma unroll
  for (int r = 0; r < 8; ++r) {
    int n = tn * 16 + r + half * 8;
    QPE[((size_t)n * 8 + h) * 512 + kcol] = (__bf16)acc[r];
  }
}

// 2b) QPE extension k = 256..511: head-masked Q (for the fused scores_e term)
__global__ void qext_kernel(const __bf16* __restrict__ Qbf, __bf16* __restrict__ QPE) {
  int n = blockIdx.x;
  int c = threadIdx.x;          // 0..255
  __bf16 qv = Qbf[n * 256 + c];
  int hc = c >> 5;
#pragma unroll
  for (int h = 0; h < 8; ++h)
    QPE[((size_t)n * 8 + h) * 512 + 256 + c] = (h == hc) ? qv : (__bf16)0.0f;
}

// 2c) qbp[h,n] = q[h,n,:] . bp_head
__global__ void qbp_kernel(const __bf16* __restrict__ Qbf, const float* __restrict__ bp,
                           float* __restrict__ qbp) {
  int t = blockIdx.x * blockDim.x + threadIdx.x;   // 0..4095
  int h = t >> 9, n = t & 511;
  float s = 0.f;
#pragma unroll
  for (int c = 0; c < 32; ++c)
    s += (float)Qbf[n * 256 + h * 32 + c] * bp[h * 32 + c];
  qbp[h * 512 + n] = s;
}

// ---------------------------------------------------------------------------
// 3) scores[h,n,m] = (scores_e + scores_p + qbp) / sqrt(32)
//    For fixed n: out[m, h] = sum_{k<512} Aext[m,k] * B[k,h]
//    Aext rows: k<256 -> embed_qk[n,m,k] (streamed NT); k>=256 -> Kbf[m,k-256]
//    B columns h<8 from QPE[n][h][k] (contiguous in k -> b128 loads); h>=8 zero
// ---------------------------------------------------------------------------
__global__ __launch_bounds__(128) void scores_kernel(
    const float* __restrict__ E,       // embed_qk (512,512,256) f32
    const __bf16* __restrict__ Kbf,
    const __bf16* __restrict__ QPE,    // [n][h][k], 512 x 8 x 512
    const float* __restrict__ qbp,
    float* __restrict__ scores)        // attn region of d_out, [h][n][m]
{
  int wid = threadIdx.x >> 5;
  int tm  = blockIdx.x * 4 + wid;      // m tile (0..31)
  int n   = blockIdx.y;                // 512
  int lane = threadIdx.x & 31, lr = lane & 15, half = lane >> 4;
  int m0 = tm * 16 + lr;
  const float*  Erow = E + ((size_t)n * 512 + m0) * 256;
  const __bf16* Bcol = QPE + ((size_t)n * 8 + lr) * 512;   // valid when lr < 8
  v8f acc = {};
  // ---- embed_qk contraction (8 chunks of 32), streaming/non-temporal ----
#pragma unroll 2
  for (int kc = 0; kc < 256; kc += 32) {
    v16bf a = cvt_a_f32_nt(Erow, kc, half);
    v16bf b = {};
    if (lr < 8) b = load_b16x16(Bcol + kc + half * 16);
    acc = WMMA_BF16(a, b, acc);
  }
  // ---- K contraction (scores_e), 8 chunks of 32 ----
  const __bf16* Krow = Kbf + (size_t)m0 * 256;
#pragma unroll 2
  for (int kc = 0; kc < 256; kc += 32) {
    v16bf a = load_a_bf16(Krow, kc, half);
    v16bf b = {};
    if (lr < 8) b = load_b16x16(Bcol + 256 + kc + half * 16);
    acc = WMMA_BF16(a, b, acc);
  }
  const float scale = 0.17677669529663687f;   // 1/sqrt(32)
  if (lr < 8) {                                // only heads 0..7 are real columns
    float qb = qbp[lr * 512 + n];
    v4f s0, s1;
#pragma unroll
    for (int r = 0; r < 4; ++r) s0[r] = (acc[r] + qb) * scale;
#pragma unroll
    for (int r = 0; r < 4; ++r) s1[r] = (acc[4 + r] + qb) * scale;
    // rows r map to consecutive m = tm*16 + half*8 + r  -> two b128 stores
    float* sp = scores + ((size_t)lr * 512 + n) * 512 + tm * 16 + half * 8;
    *(v4f*)(sp)     = s0;
    *(v4f*)(sp + 4) = s1;
  }
}

// ---------------------------------------------------------------------------
// 4) softmax over m (one wave per (h,n) row); writes f32 attn + bf16 copy
// ---------------------------------------------------------------------------
__global__ __launch_bounds__(128) void softmax_kernel(
    float* __restrict__ scores,      // [h][n][m] in-place -> attn (f32)
    __bf16* __restrict__ AttnBf)     // [h*512+n][m] bf16
{
  int wid = threadIdx.x >> 5, lane = threadIdx.x & 31;
  int row = blockIdx.x * 4 + wid;    // 0..4095
  float* s = scores + (size_t)row * 512;
  float vals[16];
  float mx = -3.402823466e38f;
#pragma unroll
  for (int i = 0; i < 16; ++i) { vals[i] = s[lane + i * 32]; mx = fmaxf(mx, vals[i]); }
#pragma unroll
  for (int off = 16; off > 0; off >>= 1) mx = fmaxf(mx, __shfl_xor(mx, off, 32));
  float sum = 0.f;
#pragma unroll
  for (int i = 0; i < 16; ++i) { vals[i] = __expf(vals[i] - mx); sum += vals[i]; }
#pragma unroll
  for (int off = 16; off > 0; off >>= 1) sum += __shfl_xor(sum, off, 32);
  float inv = 1.0f / sum;
#pragma unroll
  for (int i = 0; i < 16; ++i) {
    float a = vals[i] * inv;
    s[lane + i * 32] = a;
    AttnBf[(size_t)row * 512 + lane + i * 32] = (__bf16)a;
  }
}

// ---------------------------------------------------------------------------
// 5) hidden[n, h*32+c] = sum_m attn[h,n,m] * V[m, h*32+c]
// ---------------------------------------------------------------------------
__global__ __launch_bounds__(32) void hidden_kernel(
    const __bf16* __restrict__ AttnBf, const __bf16* __restrict__ Vbf,
    float* __restrict__ hidden)       // [n][256] region of d_out
{
  int id = blockIdx.x;                // 8 * 32 * 2
  int h = id >> 6;
  int rem = id & 63;
  int tn = rem >> 1, tc = rem & 1;
  int lane = threadIdx.x, lr = lane & 15, half = lane >> 4;
  int nrow = tn * 16 + lr;
  const __bf16* Arow = AttnBf + ((size_t)h * 512 + nrow) * 512;
  v8f acc = {};
  for (int mc = 0; mc < 512; mc += 32) {
    v16bf a = load_a_bf16(Arow, mc, half);
    v16bf b;
#pragma unroll
    for (int i = 0; i < 16; ++i)
      b[i] = Vbf[(size_t)(mc + half * 16 + i) * 256 + h * 32 + tc * 16 + lr];
    acc = WMMA_BF16(a, b, acc);
  }
#pragma unroll
  for (int r = 0; r < 8; ++r) {
    int n = tn * 16 + r + half * 8;
    hidden[(size_t)n * 256 + h * 32 + tc * 16 + lr] = acc[r];
  }
}

// ---------------------------------------------------------------------------
extern "C" void kernel_launch(void* const* d_in, const int* in_sizes, int n_in,
                              void* d_out, int out_size, void* d_ws, size_t ws_size,
                              hipStream_t stream) {
  const float* input_q = (const float*)d_in[0];
  const float* input_k = (const float*)d_in[1];
  const float* input_v = (const float*)d_in[2];
  const float* embed   = (const float*)d_in[3];
  const float* Wq = (const float*)d_in[4];
  const float* bq = (const float*)d_in[5];
  const float* Wk = (const float*)d_in[6];
  const float* bk = (const float*)d_in[7];
  const float* Wv = (const float*)d_in[8];
  const float* bv = (const float*)d_in[9];
  const float* Wp = (const float*)d_in[10];
  const float* bp = (const float*)d_in[11];
  (void)in_sizes; (void)n_in; (void)out_size; (void)ws_size;

  // workspace layout (bytes)
  char* ws = (char*)d_ws;
  __bf16* Qbf    = (__bf16*)(ws + 0);                  // 512*256*2 = 256 KB
  __bf16* Kbf    = (__bf16*)(ws + 262144);             // 256 KB
  __bf16* Vbf    = (__bf16*)(ws + 524288);             // 256 KB
  float*  qbp    = (float*) (ws + 786432);             // 16 KB
  __bf16* QPE    = (__bf16*)(ws + 802816);             // [n][h][k] 512*8*512*2 = 4 MB
  __bf16* AttnBf = (__bf16*)(ws + 802816 + 4194304);   // 4 MB

  float* hidden = (float*)d_out;                       // (512, 256)
  float* attn   = hidden + 512 * 256;                  // (8, 512, 512)

  proj_qkv_kernel<<<dim3(512, 3), 32, 0, stream>>>(input_q, input_k, input_v,
                                                   Wq, Wk, Wv, bq, bk, bv,
                                                   Qbf, Kbf, Vbf);
  qp_kernel  <<<4096, 32, 0, stream>>>(Qbf, Wp, QPE);
  qext_kernel<<<512, 256, 0, stream>>>(Qbf, QPE);
  qbp_kernel <<<16, 256, 0, stream>>>(Qbf, bp, qbp);
  scores_kernel<<<dim3(8, 512), 128, 0, stream>>>(embed, Kbf, QPE, qbp, attn);
  softmax_kernel<<<1024, 128, 0, stream>>>(attn, AttnBf);
  hidden_kernel<<<512, 32, 0, stream>>>(AttnBf, Vbf, hidden);
}
